// ConvolutionLayer_29583734735048
// MI455X (gfx1250) — matrix operand order repaired and checked by reading, not compiled, not run on hardware
//
#include <hip/hip_runtime.h>
#include <hip/hip_bf16.h>

// GCN layer fused kernel for MI455X (gfx1250, wave32, WMMA f16->f32).
// out[b,i,o] = leaky( (1/deg[b,i]) * sum_f (sum_j adj[b,i,j]*node[b,j,f]) * W[o,f] + bias[o] )

typedef _Float16 v8h  __attribute__((ext_vector_type(8)));
typedef _Float16 v16h __attribute__((ext_vector_type(16)));
typedef float    v4f  __attribute__((ext_vector_type(4)));
typedef float    v8f  __attribute__((ext_vector_type(8)));

#define NB 64
#define NN 1024
#define NF 128
#define TM 128   // rows per workgroup
#define KT 32    // K per WMMA step (f16 wmma K)
#define SLOPE 0.01f

// ---------------------------------------------------------------------------
// Prepass: nodeT[b][f][j] = (f16) node[b][j][f]   (tiled LDS transpose)
// One-time 33.5MB read + 16.7MB write; nodeT stays L2-resident (192MB L2).
// ---------------------------------------------------------------------------
__global__ __launch_bounds__(256) void node_transpose_f16_kernel(
    const float* __restrict__ node, _Float16* __restrict__ nodeT)
{
    __shared__ _Float16 t[32][34];   // +2 halves pad: conflict-free transposed reads
    const int b  = blockIdx.z;
    const int j0 = blockIdx.x * 32;
    const int f0 = blockIdx.y * 32;
    const int c  = threadIdx.x & 31;
    const int r0 = threadIdx.x >> 5;   // 0..7

    const float* src = node + ((size_t)b * NN + j0) * NF + f0;
    #pragma unroll
    for (int q = 0; q < 4; ++q) {
        const int r = r0 + q * 8;
        t[r][c] = (_Float16)src[(size_t)r * NF + c];
    }
    __syncthreads();
    _Float16* dst = nodeT + ((size_t)b * NF + f0) * NN + j0;
    #pragma unroll
    for (int q = 0; q < 4; ++q) {
        const int r = r0 + q * 8;
        dst[(size_t)r * NN + c] = t[c][r];
    }
}

// ---------------------------------------------------------------------------
// Main fused kernel. PRET=true: B tiles come pre-transposed/pre-converted
// from nodeT (coalesced f16 loads, no in-loop transpose). PRET=false:
// self-contained fallback transposing f32 node tiles through LDS.
// ---------------------------------------------------------------------------
template <bool PRET>
__global__ __launch_bounds__(256) void gcn_fused_kernel(
    const float*    __restrict__ node,    // [B,N,NF] f32
    const _Float16* __restrict__ nodeT,   // [B,NF,N] f16 (PRET only)
    const float*    __restrict__ adj,     // [B,N,N]  f32
    const float*    __restrict__ Wm,      // [NF,NF]  (out,in) f32
    const float*    __restrict__ bias,    // [NF] f32
    float*          __restrict__ out)     // [B,N,NF] f32
{
    // Phase 1: double-buffered adj/node f16 tiles (2 * (4096+4096) halves = 32KB)
    // Phase 2: reuse as agg[128][128] f16 (16384 halves = 32KB)
    __shared__ __align__(16) _Float16 smem[16384];
    __shared__ float sdeg[TM];

    const int tid  = threadIdx.x;
    const int lane = tid & 31;
    const int ln   = lane & 15;   // column-within-tile / row-within-tile
    const int hi   = lane >> 4;   // half-wave select (ISA fragment layout)
    const int w    = tid >> 5;    // wave 0..7
    const int wm   = w & 3;       // M block: rows [wm*32, wm*32+32)
    const int wf   = w >> 2;      // F block: cols [wf*64, wf*64+64)
    // wave-uniform scalar copy -> forces s_cbranch (EXEC stays all-ones at WMMAs)
    const int wfu  = __builtin_amdgcn_readfirstlane(wf);

    const int b  = blockIdx.y;
    const int i0 = blockIdx.x * TM;

    const float*    adjb   = adj  + ((size_t)b * NN + i0) * NN;
    const float*    nodeb  = node + (size_t)b * NN * NF;
    const _Float16* nodeTb = PRET ? (nodeT + (size_t)b * NF * NN) : nullptr;
    float*          outb   = out  + ((size_t)b * NN + i0) * NF;

    // f32 accumulators: 2 M-tiles x 4 N-tiles of 16x16
    v8f acc[2][4] = {};
    v8f dacc[2]   = {};           // degree (row-sum) accumulators (wf==0 waves)
    v16h bones;
    #pragma unroll
    for (int q = 0; q < 16; ++q) bones[q] = (_Float16)1.0f;
    // Pin the ones-fragment in VGPRs: without this the register allocator
    // rematerializes it (s_movs + v_movs + WAR-hazard v_nops) every iteration
    // inside the degree branch.
    asm volatile("" : "+v"(bones));

    // staging assignment (256 threads)
    const int ar = tid >> 1;         // adj local row   0..127
    const int ah = (tid & 1) * 16;   // adj col half    0 / 16
    // fallback node staging
    const int nk = tid >> 3;         // node k (j) idx  0..31
    const int nf = (tid & 7) * 16;   // node f base     0..112
    // PRET node staging: thread t loads f-row (t>>1), k-half (t&1)*16 from nodeT
    const int pf = tid >> 1;         // f row 0..127
    const int ph = (tid & 1) * 16;   // k half 0/16

    v4f areg[4];
    v4f nreg[4];     // fallback path
    v8h nhreg[2];    // PRET path (already f16)

    // prefetch step 0
    {
        const float* ap = adjb + (size_t)ar * NN + ah;
        #pragma unroll
        for (int q = 0; q < 4; ++q)
            areg[q] = __builtin_nontemporal_load((const v4f*)(ap + q * 4)); // stream adj; keep L2 for node/W
        if constexpr (PRET) {
            const v8h* np = (const v8h*)(nodeTb + (size_t)pf * NN + ph);
            nhreg[0] = np[0];
            nhreg[1] = np[1];
        } else {
            const float* np = nodeb + (size_t)nk * NF + nf;
            #pragma unroll
            for (int q = 0; q < 4; ++q) nreg[q] = *(const v4f*)(np + q * 4);
        }
    }

    const int NS = NN / KT; // 32 K-steps
    for (int s = 0; s < NS; ++s) {
        _Float16* la = smem + (s & 1) * 8192;   // adj tile  [128 i][32 k] f16
        _Float16* lb = la + 4096;               // node tile [128 f][32 k] f16

        // store staged tiles
        {
            _Float16 h16[16];
            #pragma unroll
            for (int q = 0; q < 4; ++q)
                #pragma unroll
                for (int e = 0; e < 4; ++e) h16[q * 4 + e] = (_Float16)areg[q][e];
            v8h* dst = (v8h*)&la[ar * KT + ah];
            dst[0] = *(v8h*)&h16[0];
            dst[1] = *(v8h*)&h16[8];

            if constexpr (PRET) {
                v8h* nb = (v8h*)&lb[pf * KT + ph];
                nb[0] = nhreg[0];
                nb[1] = nhreg[1];
            } else {
                // transpose on store so B fragments are contiguous K-runs
                #pragma unroll
                for (int q = 0; q < 4; ++q)
                    #pragma unroll
                    for (int e = 0; e < 4; ++e)
                        lb[(nf + q * 4 + e) * KT + nk] = (_Float16)nreg[q][e];
            }
        }
        __syncthreads();

        // prefetch next step's globals (overlaps with WMMA below)
        if (s + 1 < NS) {
            const float* ap = adjb + (size_t)ar * NN + (s + 1) * KT + ah;
            #pragma unroll
            for (int q = 0; q < 4; ++q)
                areg[q] = __builtin_nontemporal_load((const v4f*)(ap + q * 4));
            if constexpr (PRET) {
                const v8h* np = (const v8h*)(nodeTb + (size_t)pf * NN + (s + 1) * KT + ph);
                nhreg[0] = np[0];
                nhreg[1] = np[1];
            } else {
                const float* np = nodeb + (size_t)((s + 1) * KT + nk) * NF + nf;
                #pragma unroll
                for (int q = 0; q < 4; ++q) nreg[q] = *(const v4f*)(np + q * 4);
            }
        }

        // A fragments (ISA 16-bit A 16x32: lane<16 holds K 0..7,16..23; lane>=16 holds K 8..15,24..31)
        v16h afr[2];
        #pragma unroll
        for (int m = 0; m < 2; ++m) {
            const _Float16* p = &la[(wm * 32 + m * 16 + ln) * KT + hi * 8];
            v8h lo = *(const v8h*)p;
            v8h hp = *(const v8h*)(p + 16);
            #pragma unroll
            for (int e = 0; e < 8; ++e) { afr[m][e] = lo[e]; afr[m][8 + e] = hp[e]; }
        }
        // B fragments (32x16: lanes 0-15 K=0..7,16..23 of col ln, lanes 16-31 K=8..15,24..31) + WMMA
        #pragma unroll
        for (int n = 0; n < 4; ++n) {
            const _Float16* p = &lb[(wf * 64 + n * 16 + ln) * KT + hi * 8];
            v8h lo = *(const v8h*)p;
            v8h hp = *(const v8h*)(p + 16);
            v16h bfr;
            #pragma unroll
            for (int e = 0; e < 8; ++e) { bfr[e] = lo[e]; bfr[8 + e] = hp[e]; }
            #pragma unroll
            for (int m = 0; m < 2; ++m)
                acc[m][n] = __builtin_amdgcn_wmma_f32_16x16x32_f16(
                    false, afr[m], false, bfr, (short)0, acc[m][n], false, false);
        }
        // degree = adj_tile @ ones (scalar-uniform branch)
        if (wfu == 0) {
            #pragma unroll
            for (int m = 0; m < 2; ++m)
                dacc[m] = __builtin_amdgcn_wmma_f32_16x16x32_f16(
                    false, afr[m], false, bones, (short)0, dacc[m], false, false);
        }
    }

    __syncthreads();  // everyone done reading phase-1 LDS

    // spill agg (f32 acc -> f16) into smem as [row][f], row-major, stride NF
    // C layout: VGPR k <-> M = k + 8*hi, N = ln
    #pragma unroll
    for (int m = 0; m < 2; ++m)
        #pragma unroll
        for (int n = 0; n < 4; ++n)
            #pragma unroll
            for (int k = 0; k < 8; ++k)
                smem[(wm * 32 + m * 16 + k + hi * 8) * NF + wf * 64 + n * 16 + ln] =
                    (_Float16)acc[m][n][k];

    // inverse degree (all N columns of dacc hold the row sum)
    if (wfu == 0) {
        if (ln == 0) {
            #pragma unroll
            for (int m = 0; m < 2; ++m)
                #pragma unroll
                for (int k = 0; k < 8; ++k)
                    sdeg[wm * 32 + m * 16 + k + hi * 8] = 1.0f / dacc[m][k];
        }
    }
    __syncthreads();

    // Phase 2: out_tile = agg @ W^T (K = f = 128, 4 K-steps); W fragments from global (L2-hot)
    v8f acc2[2][4] = {};
    #pragma unroll
    for (int f0 = 0; f0 < NF; f0 += KT) {
        v16h afr[2];
        #pragma unroll
        for (int m = 0; m < 2; ++m) {
            const _Float16* p = &smem[(wm * 32 + m * 16 + ln) * NF + f0 + hi * 8];
            v8h lo = *(const v8h*)p;
            v8h hp = *(const v8h*)(p + 16);
            #pragma unroll
            for (int e = 0; e < 8; ++e) { afr[m][e] = lo[e]; afr[m][8 + e] = hp[e]; }
        }
        #pragma unroll
        for (int n = 0; n < 4; ++n) {
            // B[k=f][n=o] = W[o,f]: lane ln -> col o, f = hi*8+0..7 and 16+hi*8+0..7 from global row of W
            const float* wp = Wm + (size_t)(wf * 64 + n * 16 + ln) * NF + f0 + hi * 8;
            v16h bfr;
            #pragma unroll
            for (int q = 0; q < 4; ++q) {
                v4f t = *(const v4f*)(wp + (q >> 1) * 16 + (q & 1) * 4);
                #pragma unroll
                for (int e = 0; e < 4; ++e) bfr[q * 4 + e] = (_Float16)t[e];
            }
            #pragma unroll
            for (int m = 0; m < 2; ++m)
                acc2[m][n] = __builtin_amdgcn_wmma_f32_16x16x32_f16(
                    false, afr[m], false, bfr, (short)0, acc2[m][n], false, false);
        }
    }

    // epilogue: scale by 1/deg (commutes past W), + bias, LeakyReLU, NT store
    #pragma unroll
    for (int m = 0; m < 2; ++m) {
        float idg[8];
        #pragma unroll
        for (int k = 0; k < 8; ++k) idg[k] = sdeg[wm * 32 + m * 16 + k + hi * 8];
        #pragma unroll
        for (int n = 0; n < 4; ++n) {
            const int o = wf * 64 + n * 16 + ln;
            const float bv = bias[o];
            #pragma unroll
            for (int k = 0; k < 8; ++k) {
                float v = acc2[m][n][k] * idg[k] + bv;
                v = (v >= 0.0f) ? v : SLOPE * v;
                __builtin_nontemporal_store(
                    v, &outb[(size_t)(wm * 32 + m * 16 + k + hi * 8) * NF + o]);
            }
        }
    }
}

extern "C" void kernel_launch(void* const* d_in, const int* in_sizes, int n_in,
                              void* d_out, int out_size, void* d_ws, size_t ws_size,
                              hipStream_t stream) {
    (void)in_sizes; (void)n_in; (void)out_size;
    const float* node = (const float*)d_in[0];  // [64,1024,128]
    const float* adj  = (const float*)d_in[1];  // [64,1024,1024]
    const float* Wm   = (const float*)d_in[2];  // [128,128]
    const float* bias = (const float*)d_in[3];  // [128]
    float* out = (float*)d_out;                 // [64,1024,128]

    const dim3 grid(NN / TM, NB);               // 8 x 64 workgroups
    const size_t needT = (size_t)NB * NF * NN * sizeof(_Float16);  // 16.7 MB

    if (d_ws != nullptr && ws_size >= needT) {
        _Float16* nodeT = (_Float16*)d_ws;
        node_transpose_f16_kernel<<<dim3(NN / 32, NF / 32, NB), 256, 0, stream>>>(node, nodeT);
        gcn_fused_kernel<true><<<grid, 256, 0, stream>>>(node, nodeT, adj, Wm, bias, out);
    } else {
        gcn_fused_kernel<false><<<grid, 256, 0, stream>>>(node, nullptr, adj, Wm, bias, out);
    }
}
